// TopKSAE_9552007266719
// MI455X (gfx1250) — matrix operand order repaired and checked by reading, not compiled, not run on hardware
//
#include <hip/hip_runtime.h>

#define HIDDEN 2048
#define DICT   32768
#define NSAMP  4096
#define KTOP   64
#define KCAP   128

// ---------------- CDNA5 types ----------------
typedef __attribute__((ext_vector_type(16))) __bf16 v16bf;
typedef __attribute__((ext_vector_type(8)))  float  v8f;
typedef int vi4 __attribute__((vector_size(16)));   // int4 vector for async-LDS builtin

union FragAB { v16bf v; uint4 q[2]; };

// ---------------- bf16 convert (RNE) ----------------
static __device__ __forceinline__ unsigned short f2bf(float f) {
    unsigned u = __float_as_uint(f);
    unsigned r = 0x7FFFu + ((u >> 16) & 1u);
    return (unsigned short)((u + r) >> 16);
}

// ---------------- async global->LDS (CDNA5) with fallback ----------------
#if __has_builtin(__builtin_amdgcn_global_load_async_to_lds_b128) && \
    __has_builtin(__builtin_amdgcn_s_wait_asynccnt)
#define SAE_ASYNC 1
#else
#define SAE_ASYNC 0
#endif

static __device__ __forceinline__ void copy16(void* lds_dst, const void* gsrc) {
#if SAE_ASYNC
    __builtin_amdgcn_global_load_async_to_lds_b128(
        (__attribute__((address_space(1))) vi4*)(gsrc),
        (__attribute__((address_space(3))) vi4*)(lds_dst),
        0, 0);
#else
    *(uint4*)lds_dst = *(const uint4*)gsrc;
#endif
}

template <int N>
static __device__ __forceinline__ void async_wait() {
#if SAE_ASYNC
    __builtin_amdgcn_s_wait_asynccnt(N);
#endif
}

// ---------------- prepass: x fp32 -> bf16 ----------------
__global__ __launch_bounds__(256) void convert_x_kernel(
    const float* __restrict__ x, unsigned short* __restrict__ xb) {
    size_t i = (size_t)blockIdx.x * 1024 + (size_t)threadIdx.x * 4;
    float4 f = *(const float4*)(x + i);
    unsigned lo = (unsigned)f2bf(f.x) | ((unsigned)f2bf(f.y) << 16);
    unsigned hi = (unsigned)f2bf(f.z) | ((unsigned)f2bf(f.w) << 16);
    uint2 o; o.x = lo; o.y = hi;
    *(uint2*)(xb + i) = o;
}

// ---------------- prepass: W_enc [H][D] fp32 -> Wt [D][H] bf16 ----------------
__global__ __launch_bounds__(256) void transpose_wenc_kernel(
    const float* __restrict__ W, unsigned short* __restrict__ Wt) {
    __shared__ float tile[32][33];
    const int d0 = blockIdx.x * 32, h0 = blockIdx.y * 32;
    const int tx = threadIdx.x & 31, ty = threadIdx.x >> 5; // 8 row-groups
#pragma unroll
    for (int r = ty; r < 32; r += 8)
        tile[r][tx] = W[(size_t)(h0 + r) * DICT + (d0 + tx)];
    __syncthreads();
#pragma unroll
    for (int r = ty; r < 32; r += 8)
        Wt[(size_t)(d0 + r) * HIDDEN + (h0 + tx)] = f2bf(tile[tx][r]);
}

// ---------------- encode GEMM: acts = x @ W_enc + b ----------------
// Block tile 128(M) x 128(N), K-step 32, 8 wave32s: wave w -> rows [16w,16w+16),
// 8 accumulators across N (64 WMMAs per block per K-step).
// Double-buffered LDS fed by global_load_async_to_lds_b128; manual 2x unroll so
// accumulators stay pinned (no loop-phi copies) and WMMAs issue back-to-back.
#define BM 128
#define BN 128
#define BK 32
#define SUBN 8
#define KSTEPS (HIDDEN / BK)

__global__ __launch_bounds__(256, 2) void sae_encode_gemm(
    const unsigned short* __restrict__ xb,   // [NSAMP][HIDDEN] bf16
    const unsigned short* __restrict__ Wt,   // [DICT][HIDDEN]  bf16
    const float* __restrict__ b_enc,         // [DICT]
    float* __restrict__ acts)                // [NSAMP][DICT]
{
    __shared__ __align__(16) unsigned short Asm[2][BM * BK]; // 8KB x2
    __shared__ __align__(16) unsigned short Bsm[2][BN * BK]; // 8KB x2

    const int tid  = threadIdx.x;
    const int lane = tid & 31;
    const int wave = tid >> 5;
    const int half = (lane >> 4) & 1;
    const int r    = lane & 15;
    const int m0   = blockIdx.y * BM;
    const int n0   = blockIdx.x * BN;

    v8f acc[SUBN];
#pragma unroll
    for (int s = 0; s < SUBN; ++s)
        acc[s] = v8f{0.f, 0.f, 0.f, 0.f, 0.f, 0.f, 0.f, 0.f};

    auto issue = [&](int buf, int kk) {
        const int k0 = kk * BK;
        // A tile: 128 rows x 64B -> 512 16B chunks, 2 per thread
#pragma unroll
        for (int j = 0; j < 2; ++j) {
            const int c = tid + j * 256;
            const int mi = c >> 2, part = c & 3;
            copy16(&Asm[buf][mi * BK + part * 8],
                   xb + (size_t)(m0 + mi) * HIDDEN + k0 + part * 8);
        }
        // B tile: 128 rows x 64B -> 512 16B chunks, 2 per thread
#pragma unroll
        for (int j = 0; j < 2; ++j) {
            const int c = tid + j * 256;
            const int ni = c >> 2, part = c & 3;
            copy16(&Bsm[buf][ni * BK + part * 8],
                   Wt + (size_t)(n0 + ni) * HIDDEN + k0 + part * 8);
        }
    };

    auto compute = [&](int buf) {
        // A fragment: lane (half, r) holds row r, K = 8*half..+7 and 16+8*half..+7
        FragAB a;
        const unsigned short* Ab = &Asm[buf][(wave * 16 + r) * BK];
        a.q[0] = *(const uint4*)(Ab + half * 8);
        a.q[1] = *(const uint4*)(Ab + 16 + half * 8);
        // Preload all B fragments, then issue WMMAs back-to-back (no RAW chains)
        FragAB b[SUBN];
#pragma unroll
        for (int s = 0; s < SUBN; ++s) {
            const unsigned short* Bb = &Bsm[buf][(s * 16 + r) * BK];
            b[s].q[0] = *(const uint4*)(Bb + half * 8);
            b[s].q[1] = *(const uint4*)(Bb + 16 + half * 8);
        }
#pragma unroll
        for (int s = 0; s < SUBN; ++s)
            acc[s] = __builtin_amdgcn_wmma_f32_16x16x32_bf16(
                false, a.v, false, b[s].v, (short)0, acc[s], false, false);
    };

    issue(0, 0);
    for (int kk = 0; kk < KSTEPS; kk += 2) {
        // ---- stage 0: compute buffer 0, prefetch buffer 1 ----
        issue(1, kk + 1);            // kk+1 <= KSTEPS-1 always (KSTEPS even)
        async_wait<4>();             // 4 newest (buf1) may remain outstanding
        __syncthreads();
        compute(0);
        __syncthreads();
        // ---- stage 1: compute buffer 1, prefetch buffer 0 ----
        if (kk + 2 < KSTEPS) { issue(0, kk + 2); async_wait<4>(); }
        else                 { async_wait<0>(); }
        __syncthreads();
        compute(1);
        __syncthreads();
    }

    // Epilogue: D element (M = v + 8*half, N = r) per accumulator VGPR v.
    const int mrow = m0 + wave * 16 + half * 8;
#pragma unroll
    for (int s = 0; s < SUBN; ++s) {
        const int col = n0 + s * 16 + r;
        const float bias = b_enc[col];
        float* o = acts + (size_t)mrow * DICT + col;
#pragma unroll
        for (int v = 0; v < 8; ++v)
            o[(size_t)v * DICT] = acc[s][v] + bias;
    }
}

// ---------------- fused top-k select + sparsify + decode ----------------
static __device__ __forceinline__ unsigned sortkey(float f) {
    unsigned b = __float_as_uint(f);
    return (b & 0x80000000u) ? ~b : (b | 0x80000000u);
}

__global__ __launch_bounds__(256) void topk_decode_kernel(
    float* __restrict__ acts,          // [NSAMP][DICT] dense in, sparsified out
    const float* __restrict__ Wdec,    // [DICT][HIDDEN]
    float* __restrict__ recon)         // [NSAMP][HIDDEN]
{
    extern __shared__ unsigned char smem[];
    float* row      = (float*)smem;                 // DICT floats (128KB)
    int*   hist     = (int*)(row + DICT);           // 256
    int*   s_cnt    = hist + 256;                   // 256
    int*   kept_idx = s_cnt + 256;                  // KCAP
    float* kept_val = (float*)(kept_idx + KCAP);    // KCAP
    int*   s_misc   = (int*)(kept_val + KCAP);      // [0]=bin [1]=krem [2]=total

    const int n = blockIdx.x;
    const int t = threadIdx.x;
    float* arow = acts + (size_t)n * DICT;

    for (int i = t; i < DICT / 4; i += 256)
        ((float4*)row)[i] = ((const float4*)arow)[i];
    __syncthreads();

    // 4-pass MSB-first radix select for the exact K-th largest key
    unsigned prefix = 0;
    int krem = KTOP;
    for (int level = 0; level < 4; ++level) {
        const int shift = 24 - 8 * level;
        const unsigned pmask = (level == 0) ? 0u : (0xFFFFFFFFu << (shift + 8));
        for (int b = t; b < 256; b += 256) hist[b] = 0;
        __syncthreads();
        for (int i = t; i < DICT; i += 256) {
            unsigned u = sortkey(row[i]);
            if ((u & pmask) == prefix) atomicAdd(&hist[(u >> shift) & 255], 1);
        }
        __syncthreads();
        if (t == 0) {
            int cum = 0, b = 255;
            for (; b > 0; --b) {
                if (cum + hist[b] >= krem) break;
                cum += hist[b];
            }
            s_misc[0] = b;
            s_misc[1] = krem - cum;
        }
        __syncthreads();
        prefix |= ((unsigned)s_misc[0]) << shift;
        krem = s_misc[1];
        __syncthreads();
    }
    const unsigned thresh = prefix; // key of K-th largest element

    // Deterministic index-ordered compaction (blocked ranges + exclusive scan)
    const int per = DICT / 256; // 128
    const int base = t * per;
    int cnt = 0;
    for (int i = base; i < base + per; ++i) cnt += (sortkey(row[i]) >= thresh);
    s_cnt[t] = cnt;
    __syncthreads();
    if (t == 0) {
        int run = 0;
        for (int i = 0; i < 256; ++i) { int c = s_cnt[i]; s_cnt[i] = run; run += c; }
        s_misc[2] = run;
    }
    __syncthreads();
    int pos = s_cnt[t];
    for (int i = base; i < base + per; ++i) {
        float v = row[i];
        if (sortkey(v) >= thresh && pos < KCAP) { kept_idx[pos] = i; kept_val[pos] = v; ++pos; }
    }
    __syncthreads();
    const int total = min(s_misc[2], KCAP);

    // Decode: thread t accumulates output columns [8t, 8t+8) in registers
    float accv[8];
#pragma unroll
    for (int i = 0; i < 8; ++i) accv[i] = 0.f;
    for (int j = 0; j < total; ++j) {
        const float v = kept_val[j];
        const float* wrow = Wdec + (size_t)kept_idx[j] * HIDDEN + t * 8;
        float4 w0 = ((const float4*)wrow)[0];
        float4 w1 = ((const float4*)wrow)[1];
        accv[0] += v * w0.x; accv[1] += v * w0.y; accv[2] += v * w0.z; accv[3] += v * w0.w;
        accv[4] += v * w1.x; accv[5] += v * w1.y; accv[6] += v * w1.z; accv[7] += v * w1.w;
    }
    float* rrow = recon + (size_t)n * HIDDEN + t * 8;
    float4 o0, o1;
    o0.x = accv[0]; o0.y = accv[1]; o0.z = accv[2]; o0.w = accv[3];
    o1.x = accv[4]; o1.y = accv[5]; o1.z = accv[6]; o1.w = accv[7];
    ((float4*)rrow)[0] = o0;
    ((float4*)rrow)[1] = o1;

    // In-place sparsification (coalesced)
    for (int i = t; i < DICT; i += 256) {
        float v = row[i];
        arow[i] = (sortkey(v) >= thresh) ? v : 0.0f;
    }
}

// ---------------- host launcher ----------------
extern "C" void kernel_launch(void* const* d_in, const int* in_sizes, int n_in,
                              void* d_out, int out_size, void* d_ws, size_t ws_size,
                              hipStream_t stream) {
    (void)in_sizes; (void)n_in; (void)out_size; (void)ws_size;
    const float* x     = (const float*)d_in[0];
    const float* W_enc = (const float*)d_in[1];
    const float* b_enc = (const float*)d_in[2];
    const float* W_dec = (const float*)d_in[3];

    float* recon  = (float*)d_out;                         // [NSAMP][HIDDEN]
    float* sparse = recon + (size_t)NSAMP * HIDDEN;        // [NSAMP][DICT]

    unsigned short* Wt = (unsigned short*)d_ws;                    // 128MB
    unsigned short* xb = Wt + (size_t)DICT * HIDDEN;               // +16MB

    convert_x_kernel<<<(NSAMP * HIDDEN) / 1024, 256, 0, stream>>>(x, xb);
    transpose_wenc_kernel<<<dim3(DICT / 32, HIDDEN / 32), 256, 0, stream>>>(W_enc, Wt);
    sae_encode_gemm<<<dim3(DICT / BN, NSAMP / BM), 256, 0, stream>>>(xb, Wt, b_enc, sparse);

    const size_t smem = (size_t)(DICT + 256 + 256 + KCAP) * 4 + KCAP * 4 + 16;
    topk_decode_kernel<<<NSAMP, 256, smem, stream>>>(sparse, W_dec, recon);
}
